// GIN_26096221290965
// MI455X (gfx1250) — matrix operand order repaired and checked by reading, not compiled, not run on hardware
//
#include <hip/hip_runtime.h>
#include <hip/hip_bf16.h>

#define N_NODES_C  50000
#define N_EDGES_C  800000
#define DIM        128
#define N_GRAPHS_C 512
#define TM         32   // node rows per block in GEMM kernels

typedef __bf16 bf16;
typedef __attribute__((ext_vector_type(16))) __bf16 v16bf;
typedef __attribute__((ext_vector_type(8)))  __bf16 v8bf;
typedef __attribute__((ext_vector_type(8)))  float  v8f;

// float -> bf16 with round-to-nearest-even (bit-exact, no reliance on cast codegen)
__device__ inline bf16 f2bf(float f) {
    union { float f; unsigned u; } c; c.f = f;
    unsigned u = c.u;
    unsigned r = (u + 0x7FFFu + ((u >> 16) & 1u)) >> 16;
    unsigned short s = (unsigned short)r;
    bf16 b;
    __builtin_memcpy(&b, &s, 2);
    return b;
}

// A fragment: 16x32 bf16 tile, row-major source with leading dim ld.
// ISA layout: lane L (L<16) row=L holds K = kb*32 + {0..7, 16..23};
//             lane L+16 row=L holds K = kb*32 + {8..15, 24..31}.
__device__ inline v16bf load_a_frag(const bf16* Z, int ld, int lane, int kb) {
    int row = lane & 15;
    int k0  = kb * 32 + ((lane & 16) ? 8 : 0);
    const bf16* p = Z + row * ld + k0;
    v8bf lo = *(const v8bf*)(p);
    v8bf hi = *(const v8bf*)(p + 16);
    return __builtin_shufflevector(lo, hi, 0,1,2,3,4,5,6,7,8,9,10,11,12,13,14,15);
}

// B fragment: 32x16 bf16 tile from TRANSPOSED weights WT[n][k] (leading dim ld).
// ISA layout: lane L holds col N = L&15, K = kb*32 + (L&16 ? {8..15, 24..31} : {0..7, 16..23}).
__device__ inline v16bf load_b_frag(const bf16* WT, int ld, int lane, int kb, int nb) {
    int col = nb * 16 + (lane & 15);
    int k0  = kb * 32 + ((lane & 16) ? 8 : 0);
    const bf16* p = WT + col * ld + k0;
    v8bf lo = *(const v8bf*)(p);
    v8bf hi = *(const v8bf*)(p + 16);
    return __builtin_shufflevector(lo, hi, 0,1,2,3,4,5,6,7,8,9,10,11,12,13,14,15);
}

__global__ void zero_kernel(float* __restrict__ p, int n) {
    int i = blockIdx.x * blockDim.x + threadIdx.x;
    if (i < n) p[i] = 0.0f;
}

// One wave (32 lanes) per edge; each lane moves 4 floats (float4) => 128 features.
__global__ void agg_kernel(const float* __restrict__ h,
                           const int* __restrict__ src,
                           const int* __restrict__ dst,
                           float* __restrict__ agg, int n_edges) {
    int e    = (blockIdx.x * blockDim.x + threadIdx.x) >> 5;
    int lane = threadIdx.x & 31;
    if (e >= n_edges) return;
    int s = src[e];
    int d = dst[e];
    float4 v = ((const float4*)(h + (size_t)s * DIM))[lane];
    float* dp = agg + (size_t)d * DIM + lane * 4;
    atomicAdd(dp + 0, v.x);
    atomicAdd(dp + 1, v.y);
    atomicAdd(dp + 2, v.z);
    atomicAdd(dp + 3, v.w);
}

// One wave per node: g[batch[node]] += h[node]
__global__ void pool_kernel(const float* __restrict__ h,
                            const int* __restrict__ batch,
                            float* __restrict__ g, int n_nodes) {
    int node = (blockIdx.x * blockDim.x + threadIdx.x) >> 5;
    int lane = threadIdx.x & 31;
    if (node >= n_nodes) return;
    int b = batch[node];
    float4 v = ((const float4*)(h + (size_t)node * DIM))[lane];
    float* dp = g + (size_t)b * DIM + lane * 4;
    atomicAdd(dp + 0, v.x);
    atomicAdd(dp + 1, v.y);
    atomicAdd(dp + 2, v.z);
    atomicAdd(dp + 3, v.w);
}

// Fused GIN MLP: out = relu( relu((x+agg) @ Wa + ba) @ Wb + bb ) for a TM-row tile.
// 8 waves; wave w owns output column tile nb=w, row tiles rt=0..1.
__global__ __launch_bounds__(256)
void gin_mlp_kernel(const float* __restrict__ x, const float* __restrict__ agg,
                    const float* __restrict__ Wa, const float* __restrict__ ba,
                    const float* __restrict__ Wb, const float* __restrict__ bb,
                    float* __restrict__ out, int n_rows) {
    __shared__ __attribute__((aligned(64))) bf16 sZ[TM * DIM];   // 8 KB
    __shared__ __attribute__((aligned(64))) bf16 sW[DIM * DIM];  // 32 KB, transposed [n][k]
    __shared__ __attribute__((aligned(64))) bf16 sH[TM * DIM];   // 8 KB

    const int tid  = threadIdx.x;
    const int lane = tid & 31;
    const int wave = tid >> 5;                 // output column tile
    const int rowBase = blockIdx.x * TM;

    // z = x + agg, converted to bf16 (zero pad past n_rows)
    for (int i = tid; i < TM * DIM; i += 256) {
        int r = i >> 7, c = i & (DIM - 1);
        int node = rowBase + r;
        float v = 0.0f;
        if (node < n_rows) v = x[(size_t)node * DIM + c] + agg[(size_t)node * DIM + c];
        sZ[i] = f2bf(v);
    }
    // Wa transposed into LDS: sW[n*DIM + k] = Wa[k*DIM + n]
    for (int i = tid; i < DIM * DIM; i += 256) {
        int k = i >> 7, n = i & (DIM - 1);
        sW[n * DIM + k] = f2bf(Wa[i]);
    }
    __syncthreads();

    const int n   = lane & 15;
    const int mhi = (lane & 16) ? 8 : 0;

    // ---- GEMM1: H = relu(Z @ Wa + ba) ----
    {
        v8f acc0 = {}; v8f acc1 = {};
        for (int kb = 0; kb < 4; ++kb) {
            v16bf bf = load_b_frag(sW, DIM, lane, kb, wave);
            v16bf a0 = load_a_frag(sZ,            DIM, lane, kb);
            v16bf a1 = load_a_frag(sZ + 16 * DIM, DIM, lane, kb);
            acc0 = __builtin_amdgcn_wmma_f32_16x16x32_bf16(false, a0, false, bf, (short)0, acc0, false, false);
            acc1 = __builtin_amdgcn_wmma_f32_16x16x32_bf16(false, a1, false, bf, (short)0, acc1, false, false);
        }
        float bias = ba[wave * 16 + n];
        for (int e = 0; e < 8; ++e) {
            float v0 = acc0[e] + bias; v0 = v0 > 0.0f ? v0 : 0.0f;
            float v1 = acc1[e] + bias; v1 = v1 > 0.0f ? v1 : 0.0f;
            sH[(mhi + e) * DIM + wave * 16 + n]      = f2bf(v0);
            sH[(16 + mhi + e) * DIM + wave * 16 + n] = f2bf(v1);
        }
    }
    __syncthreads();

    // Wb transposed into LDS (reuse sW)
    for (int i = tid; i < DIM * DIM; i += 256) {
        int k = i >> 7, nn = i & (DIM - 1);
        sW[nn * DIM + k] = f2bf(Wb[i]);
    }
    __syncthreads();

    // ---- GEMM2: out = relu(H @ Wb + bb) ----
    {
        v8f acc0 = {}; v8f acc1 = {};
        for (int kb = 0; kb < 4; ++kb) {
            v16bf bf = load_b_frag(sW, DIM, lane, kb, wave);
            v16bf a0 = load_a_frag(sH,            DIM, lane, kb);
            v16bf a1 = load_a_frag(sH + 16 * DIM, DIM, lane, kb);
            acc0 = __builtin_amdgcn_wmma_f32_16x16x32_bf16(false, a0, false, bf, (short)0, acc0, false, false);
            acc1 = __builtin_amdgcn_wmma_f32_16x16x32_bf16(false, a1, false, bf, (short)0, acc1, false, false);
        }
        float bias = bb[wave * 16 + n];
        for (int e = 0; e < 8; ++e) {
            int r0 = rowBase + mhi + e;
            int r1 = rowBase + 16 + mhi + e;
            float v0 = acc0[e] + bias; v0 = v0 > 0.0f ? v0 : 0.0f;
            float v1 = acc1[e] + bias; v1 = v1 > 0.0f ? v1 : 0.0f;
            if (r0 < n_rows) out[(size_t)r0 * DIM + wave * 16 + n] = v0;
            if (r1 < n_rows) out[(size_t)r1 * DIM + wave * 16 + n] = v1;
        }
    }
}

// Final FC: out = relu(g @ Wfc + bfc), one GEMM per TM-row tile.
__global__ __launch_bounds__(256)
void fc_kernel(const float* __restrict__ g, const float* __restrict__ Wfc,
               const float* __restrict__ bfc, float* __restrict__ out, int n_rows) {
    __shared__ __attribute__((aligned(64))) bf16 sZ[TM * DIM];
    __shared__ __attribute__((aligned(64))) bf16 sW[DIM * DIM];

    const int tid  = threadIdx.x;
    const int lane = tid & 31;
    const int wave = tid >> 5;
    const int rowBase = blockIdx.x * TM;

    for (int i = tid; i < TM * DIM; i += 256) {
        int r = i >> 7, c = i & (DIM - 1);
        int row = rowBase + r;
        float v = (row < n_rows) ? g[(size_t)row * DIM + c] : 0.0f;
        sZ[i] = f2bf(v);
    }
    for (int i = tid; i < DIM * DIM; i += 256) {
        int k = i >> 7, nn = i & (DIM - 1);
        sW[nn * DIM + k] = f2bf(Wfc[i]);
    }
    __syncthreads();

    const int n   = lane & 15;
    const int mhi = (lane & 16) ? 8 : 0;

    v8f acc0 = {}; v8f acc1 = {};
    for (int kb = 0; kb < 4; ++kb) {
        v16bf bf = load_b_frag(sW, DIM, lane, kb, wave);
        v16bf a0 = load_a_frag(sZ,            DIM, lane, kb);
        v16bf a1 = load_a_frag(sZ + 16 * DIM, DIM, lane, kb);
        acc0 = __builtin_amdgcn_wmma_f32_16x16x32_bf16(false, a0, false, bf, (short)0, acc0, false, false);
        acc1 = __builtin_amdgcn_wmma_f32_16x16x32_bf16(false, a1, false, bf, (short)0, acc1, false, false);
    }
    float bias = bfc[wave * 16 + n];
    for (int e = 0; e < 8; ++e) {
        int r0 = rowBase + mhi + e;
        int r1 = rowBase + 16 + mhi + e;
        float v0 = acc0[e] + bias; v0 = v0 > 0.0f ? v0 : 0.0f;
        float v1 = acc1[e] + bias; v1 = v1 > 0.0f ? v1 : 0.0f;
        if (r0 < n_rows) out[(size_t)r0 * DIM + wave * 16 + n] = v0;
        if (r1 < n_rows) out[(size_t)r1 * DIM + wave * 16 + n] = v1;
    }
}

extern "C" void kernel_launch(void* const* d_in, const int* in_sizes, int n_in,
                              void* d_out, int out_size, void* d_ws, size_t ws_size,
                              hipStream_t stream) {
    (void)in_sizes; (void)n_in; (void)out_size; (void)ws_size;

    const float* x    = (const float*)d_in[0];
    const int*   ei   = (const int*)d_in[1];
    const int*   bat  = (const int*)d_in[2];
    const float* W1a  = (const float*)d_in[3];
    const float* b1a  = (const float*)d_in[4];
    const float* W1b  = (const float*)d_in[5];
    const float* b1b  = (const float*)d_in[6];
    const float* W2a  = (const float*)d_in[7];
    const float* b2a  = (const float*)d_in[8];
    const float* W2b  = (const float*)d_in[9];
    const float* b2b  = (const float*)d_in[10];
    const float* Wfc  = (const float*)d_in[11];
    const float* bfc  = (const float*)d_in[12];
    float* out = (float*)d_out;

    const int feat_elems = N_NODES_C * DIM;          // 6.4M floats
    float* agg = (float*)d_ws;
    float* h1  = agg + (size_t)feat_elems;
    float* h2  = h1 + (size_t)feat_elems;
    float* g   = h2 + (size_t)feat_elems;            // 512*128

    const int* src = ei;
    const int* dst = ei + N_EDGES_C;

    const int zb    = (feat_elems + 255) / 256;
    const int eb    = (N_EDGES_C * 32 + 255) / 256;
    const int gemmb = (N_NODES_C + TM - 1) / TM;
    const int pb    = (N_NODES_C * 32 + 255) / 256;
    const int gzb   = (N_GRAPHS_C * DIM + 255) / 256;

    // Layer 1
    zero_kernel<<<zb, 256, 0, stream>>>(agg, feat_elems);
    agg_kernel<<<eb, 256, 0, stream>>>(x, src, dst, agg, N_EDGES_C);
    gin_mlp_kernel<<<gemmb, 256, 0, stream>>>(x, agg, W1a, b1a, W1b, b1b, h1, N_NODES_C);

    // Layer 2
    zero_kernel<<<zb, 256, 0, stream>>>(agg, feat_elems);
    agg_kernel<<<eb, 256, 0, stream>>>(h1, src, dst, agg, N_EDGES_C);
    gin_mlp_kernel<<<gemmb, 256, 0, stream>>>(h1, agg, W2a, b2a, W2b, b2b, h2, N_NODES_C);

    // Pool + FC
    zero_kernel<<<gzb, 256, 0, stream>>>(g, N_GRAPHS_C * DIM);
    pool_kernel<<<pb, 256, 0, stream>>>(h2, bat, g, N_NODES_C);
    fc_kernel<<<N_GRAPHS_C / TM, 256, 0, stream>>>(g, Wfc, bfc, out, N_GRAPHS_C);
}
